// conv2d_17145509445976
// MI455X (gfx1250) — matrix operand order repaired and checked by reading, not compiled, not run on hardware
//
#include <hip/hip_runtime.h>
#include <hip/hip_bf16.h>

// ---------------- CDNA5 WMMA types ----------------
typedef __attribute__((ext_vector_type(16))) __bf16 v16bf;
typedef __attribute__((ext_vector_type(8)))  float  v8f;

union Frag16 { uint4 q[2]; v16bf v; };

// Shapes (fixed by reference setup_inputs)
#define Bsz 16
#define Csz 64
#define Hsz 64
#define Wsz 64
#define Osz 64
#define KK  3
#define EPSV 1e-6f
#define SLOPE 0.01f

// workspace layout (bytes)
#define WPACK_OFF  ((size_t)0)                        // 2*9*4*32*16 bf16 = 73728 B
#define XG_OFF     ((size_t)131072)                   // [B][2][H][W][32] bf16 = 8 MB
#define XG_SZ      ((size_t)Bsz*2*Hsz*Wsz*32*2)
#define CMAX_OFF   (XG_OFF + XG_SZ)
#define CSZ        ((size_t)Bsz*2*Hsz*Wsz*4)          // 512 KB each
#define CMIN_OFF   (CMAX_OFF + CSZ)
#define STDS_OFF   (CMIN_OFF + CSZ)

static __device__ __forceinline__ unsigned short f2bf(float f) {
    unsigned int u = __float_as_uint(f);
    u += 0x7FFFu + ((u >> 16) & 1u);   // round-to-nearest-even
    return (unsigned short)(u >> 16);
}

// ---------------------------------------------------------------------------
// Kernel 0: pack weights fp32 -> bf16 in exact A-fragment lane order.
// A matrix = weights, 16(M=o) x 32(K=ch of one parity), contiguous k order, same as B fragment:
//   lane<16 : elems e -> K = e
//   lane>=16: elems e -> K = 16 + e
// Flat index matches main kernel: (((p*9+s)*4+mt)*32+lane)*16+e
// value = W[o, c=2*ch+p, j, i]  with s = i*3+j.
// ---------------------------------------------------------------------------
__global__ void pack_weights_k(const float* __restrict__ W,
                               unsigned short* __restrict__ wpack) {
    int idx = blockIdx.x * 256 + threadIdx.x;
    if (idx >= 2 * 9 * 4 * 32 * 16) return;
    int e    = idx & 15;
    int lane = (idx >> 4) & 31;
    int mt   = (idx >> 9) & 3;
    int rest = idx >> 11;            // p*9 + s
    int s = rest % 9, p = rest / 9;
    int i = s / 3, j = s % 3;
    int o  = mt * 16 + (lane & 15);
    int ch = (lane >> 4) * 16 + e;
    int c  = ch * 2 + p;
    float v = W[(((size_t)o * Csz + c) * KK + j) * KK + i];
    wpack[idx] = f2bf(v);
}

// ---------------------------------------------------------------------------
// Kernel 1: transpose input to [b][p][h][w][32ch] bf16 and per-pixel
// per-parity channel max/min.
// ---------------------------------------------------------------------------
__global__ void prep_input_k(const float* __restrict__ xx,
                             unsigned short* __restrict__ xg,
                             float* __restrict__ cmax,
                             float* __restrict__ cmin) {
    int idx = blockIdx.x * 256 + threadIdx.x;  // b*h*w
    if (idx >= Bsz * Hsz * Wsz) return;
    int w = idx & 63, h = (idx >> 6) & 63, b = idx >> 12;
    float mx0 = -1e30f, mx1 = -1e30f, mn0 = 1e30f, mn1 = 1e30f;
    for (int c = 0; c < Csz; ++c) {
        float v = xx[(((size_t)b * Csz + c) * Hsz + h) * Wsz + w];
        int p = c & 1, ch = c >> 1;
        if (p) { mx1 = fmaxf(mx1, v); mn1 = fminf(mn1, v); }
        else   { mx0 = fmaxf(mx0, v); mn0 = fminf(mn0, v); }
        xg[((((size_t)b * 2 + p) * Hsz + h) * Wsz + w) * 32 + ch] = f2bf(v);
    }
    size_t o0 = (((size_t)b * 2 + 0) * Hsz + h) * Wsz + w;
    size_t o1 = (((size_t)b * 2 + 1) * Hsz + h) * Wsz + w;
    cmax[o0] = mx0; cmin[o0] = mn0;
    cmax[o1] = mx1; cmin[o1] = mn1;
}

// ---------------------------------------------------------------------------
// Kernel 2: 3x3 clamped-window max/min -> stds = max - min.
// ---------------------------------------------------------------------------
__global__ void window_minmax_k(const float* __restrict__ cmax,
                                const float* __restrict__ cmin,
                                float* __restrict__ stds) {
    int idx = blockIdx.x * 256 + threadIdx.x;  // b*p*h*w
    if (idx >= Bsz * 2 * Hsz * Wsz) return;
    int w = idx & 63, h = (idx >> 6) & 63;
    int bp = idx >> 12;                       // b*2+p
    float mx = -1e30f, mn = 1e30f;
    for (int di = -1; di <= 1; ++di) {
        int hh = min(max(h + di, 0), Hsz - 1);
        for (int dj = -1; dj <= 1; ++dj) {
            int ww = min(max(w + dj, 0), Wsz - 1);
            size_t a = ((size_t)bp * Hsz + hh) * Wsz + ww;
            mx = fmaxf(mx, cmax[a]);
            mn = fminf(mn, cmin[a]);
        }
    }
    stds[idx] = mx - mn;
}

// ---------------------------------------------------------------------------
// Kernel 3: main implicit-GEMM conv with BF16 WMMA.
//   1 workgroup (256 thr = 8 waves) per (b, h). M = o (64), N = w (64),
//   K = 32ch x 9 taps x 2 parities (separate accumulators per parity).
//   Wave wv: w-tile nt = wv&3, o-tiles mt0=(wv>>2)*2, mt0+1.
//   LDS tile staged via CDNA5 async-to-LDS engine (ASYNCcnt), no VGPR
//   round-trip: global_load_async_to_lds_b128 + s_wait_asynccnt.
//   B fragment (patches, 32x16, K=ch contiguous per lane) from LDS.
//   A fragment (weights) pre-packed in global, 2 x 16B loads per lane.
// ---------------------------------------------------------------------------
__global__ __launch_bounds__(256) void conv_main_k(
        const unsigned short* __restrict__ xg,
        const unsigned short* __restrict__ wpack,
        const float* __restrict__ stds,
        const float* __restrict__ bias,
        float* __restrict__ out) {
    int b = blockIdx.x >> 6;
    int h = blockIdx.x & 63;

    // [parity][row(3)][col(66)][ch(32)] bf16  = 25344 B
    __shared__ __align__(16) unsigned short xlds[2 * 3 * 66 * 32];
    __shared__ float sst[2][64];
    __shared__ float sbias[64];

    // ---- stage input rows h-1..h+1 (replicate-clamped), both parities ----
    // 16B chunks: 2*3*66*4 = 1584, copied global->LDS by the async engine.
    for (int cid = threadIdx.x; cid < 1584; cid += 256) {
        int p    = cid / 792;
        int rem  = cid % 792;
        int i    = rem / 264;
        int rem2 = rem % 264;
        int col  = rem2 >> 2;
        int ch0  = (rem2 & 3) * 8;
        int hs = min(max(h - 1 + i, 0), Hsz - 1);
        int ws = min(max(col - 1, 0), Wsz - 1);
        const unsigned short* src =
            xg + (((((size_t)b * 2 + p) * Hsz + hs) * Wsz + ws) * 32 + ch0);
        // low 32 bits of a generic shared pointer == LDS byte address
        unsigned lds_addr = (unsigned)(unsigned long long)(
            xlds + (((p * 3 + i) * 66 + col) * 32 + ch0));
        asm volatile("global_load_async_to_lds_b128 %0, %1, off"
                     :: "v"(lds_addr), "v"(src) : "memory");
    }
    if (threadIdx.x < 128) {
        int p = threadIdx.x >> 6, w = threadIdx.x & 63;
        sst[p][w] = stds[((((size_t)b * 2 + p) * Hsz + h) * Wsz) + w];
    }
    if (threadIdx.x < 64) sbias[threadIdx.x] = bias[threadIdx.x];
    asm volatile("s_wait_asynccnt 0x0" ::: "memory");
    __syncthreads();

    int wv = threadIdx.x >> 5, lane = threadIdx.x & 31;
    int lo16 = lane & 15, hi = lane >> 4;
    int nt = wv & 3;            // w tile
    int mt0 = (wv >> 2) * 2;    // o tiles: mt0, mt0+1

    v8f acc00 = {}, acc01 = {}, acc10 = {}, acc11 = {}; // [m][parity]
    const uint4* xq = reinterpret_cast<const uint4*>(xlds);
    const uint4* wq = reinterpret_cast<const uint4*>(wpack);

#pragma unroll
    for (int s = 0; s < 9; ++s) {
        int i = s / 3, j = s % 3;
        // prefetch next K-step's weight fragments into L2/L0
        if (s < 8) {
            __builtin_prefetch(&wq[(size_t)(((0 * 9 + s + 1) * 4 + mt0) * 32 + lane) * 2], 0, 1);
        }
#pragma unroll
        for (int p = 0; p < 2; ++p) {
            // B fragment: patches, K = ch 16 contiguous per lane half
            int col = nt * 16 + lo16 + j;
            int elem = ((p * 3 + i) * 66 + col) * 32 + hi * 16; // ushort idx, 32B-aligned
            Frag16 bf;
            bf.q[0] = xq[elem >> 3];
            bf.q[1] = xq[(elem >> 3) + 1];
            // A fragments: pre-packed weights
            Frag16 a0, a1;
            size_t q0 = (size_t)((((p * 9 + s) * 4 + mt0) * 32) + lane) * 2;
            size_t q1 = (size_t)((((p * 9 + s) * 4 + mt0 + 1) * 32) + lane) * 2;
            a0.q[0] = wq[q0]; a0.q[1] = wq[q0 + 1];
            a1.q[0] = wq[q1]; a1.q[1] = wq[q1 + 1];
            if (p == 0) {
                acc00 = __builtin_amdgcn_wmma_f32_16x16x32_bf16(
                    false, a0.v, false, bf.v, (short)0, acc00, false, false);
                acc10 = __builtin_amdgcn_wmma_f32_16x16x32_bf16(
                    false, a1.v, false, bf.v, (short)0, acc10, false, false);
            } else {
                acc01 = __builtin_amdgcn_wmma_f32_16x16x32_bf16(
                    false, a0.v, false, bf.v, (short)0, acc01, false, false);
                acc11 = __builtin_amdgcn_wmma_f32_16x16x32_bf16(
                    false, a1.v, false, bf.v, (short)0, acc11, false, false);
            }
        }
    }

    // ---- epilogue: combine parities, bias, leaky-relu, rescale, store ----
    int w = nt * 16 + lo16;      // N = lane&15 (D layout)
    float s0 = sst[0][w], s1 = sst[1][w];
    float inv0 = 1.0f / (s0 + EPSV), inv1 = 1.0f / (s1 + EPSV);
    float sc0 = s0 + EPSV, sc1 = s1 + EPSV;
#pragma unroll
    for (int m = 0; m < 2; ++m) {
        v8f ae = (m == 0) ? acc00 : acc10;
        v8f ao = (m == 0) ? acc01 : acc11;
#pragma unroll
        for (int r = 0; r < 8; ++r) {
            int o = (mt0 + m) * 16 + hi * 8 + r;   // M = r + 8*(lane>=16)
            float v = ae[r] * inv0 + ao[r] * inv1 + sbias[o];
            v = (v >= 0.0f) ? v : SLOPE * v;
            v *= (r & 1) ? sc1 : sc0;              // (o&1) == (r&1)
            out[(((size_t)b * Osz + o) * Hsz + h) * Wsz + w] = v;
        }
    }
}

extern "C" void kernel_launch(void* const* d_in, const int* in_sizes, int n_in,
                              void* d_out, int out_size, void* d_ws, size_t ws_size,
                              hipStream_t stream) {
    const float* xx = (const float*)d_in[0];
    const float* W  = (const float*)d_in[1];
    const float* bi = (const float*)d_in[2];
    float* out = (float*)d_out;

    char* ws = (char*)d_ws;
    unsigned short* wpack = (unsigned short*)(ws + WPACK_OFF);
    unsigned short* xg    = (unsigned short*)(ws + XG_OFF);
    float* cmax = (float*)(ws + CMAX_OFF);
    float* cmin = (float*)(ws + CMIN_OFF);
    float* stds = (float*)(ws + STDS_OFF);

    pack_weights_k<<<144, 256, 0, stream>>>(W, wpack);
    prep_input_k<<<Bsz * Hsz * Wsz / 256, 256, 0, stream>>>(xx, xg, cmax, cmin);
    window_minmax_k<<<Bsz * 2 * Hsz * Wsz / 256, 256, 0, stream>>>(cmax, cmin, stds);
    conv_main_k<<<Bsz * Hsz, 256, 0, stream>>>(xg, wpack, stds, bi, out);
}
